// SingleLIFBlock_5196910428195
// MI455X (gfx1250) — matrix operand order repaired and checked by reading, not compiled, not run on hardware
//
#include <hip/hip_runtime.h>
#include <stdint.h>

// ---------------------------------------------------------------------------
// SNN block: out = LIF(x @ W1^T + b1) @ W2^T + b2     (T=64, B=1024, N=512)
// 3-phase pipeline, all GEMM work on v_wmma_f32_16x16x32_bf16 (wave32 CDNA5).
// Round 2: preload 8 A-fragments per K-tile (one ds clause -> 16 wmma chain),
//          CH=128 chunks to halve barrier count.
// ---------------------------------------------------------------------------

typedef __attribute__((ext_vector_type(16))) __bf16 v16bf;
typedef __attribute__((ext_vector_type(8)))  float  v8f;

struct __align__(8) US4 { unsigned short x, y, z, w; };

static constexpr int TT   = 64;
static constexpr int BB   = 1024;
static constexpr int NN   = 512;
static constexpr int KK   = 512;
static constexpr int MROW = TT * BB;      // 65536 GEMM rows
static constexpr int ROWS = 128;          // rows per workgroup
static constexpr int CH   = 128;          // K-chunk staged in LDS
static constexpr int LDA  = CH + 8;       // 136 bf16 = 272B row stride (16B aligned)

__device__ __forceinline__ unsigned short f2bf(float f) {
    union { float f; uint32_t u; } c; c.f = f;
    uint32_t u = c.u;
    uint32_t r = (u + 0x7FFFu + ((u >> 16) & 1u)) >> 16;   // round-to-nearest-even
    return (unsigned short)r;
}
__device__ __forceinline__ float bf2f(unsigned short h) {
    union { uint32_t u; float f; } c; c.u = ((uint32_t)h) << 16;
    return c.f;
}

// ------------------------- weight fp32 -> bf16 prep -------------------------
__global__ void cvt_weights(const float* __restrict__ w1, const float* __restrict__ w2,
                            unsigned short* __restrict__ o1, unsigned short* __restrict__ o2) {
    int i = blockIdx.x * blockDim.x + threadIdx.x;   // grid covers 512*512
    o1[i] = f2bf(w1[i]);
    o2[i] = f2bf(w2[i]);
}

// --------------------- WMMA GEMM:  Out = A @ W^T + bias ---------------------
// A: [65536][512] (fp32 x, or bf16 spikes).  W: [512][512] bf16 row-major
// (row m = output neuron, col k) == exactly the B-fragment's N-major layout.
// Out: bf16 h workspace (phase 1) or fp32 d_out (phase 3).
template <bool A_F32, bool OUT_F32>
__global__ __launch_bounds__(512, 1) void snn_gemm(
    const void* __restrict__ Aptr,
    const unsigned short* __restrict__ W,
    const float* __restrict__ bias,
    void* __restrict__ Out)
{
    __shared__ unsigned short As[ROWS * LDA];

    const int tid  = threadIdx.x;
    const int lane = tid & 31;
    const int wave = tid >> 5;           // 16 waves
    const int l    = lane & 15;          // N / M sub-index within half-wave
    const int hi   = lane >> 4;          // half-wave select
    const int row0 = blockIdx.x * ROWS;
    const int colb = wave * 32;          // 16 waves x 32 cols = 512

    const float b0 = bias[colb + l];
    const float b1 = bias[colb + 16 + l];

    v8f acc[8][2];
#pragma unroll
    for (int mt = 0; mt < 8; ++mt)
#pragma unroll
        for (int i = 0; i < 8; ++i) { acc[mt][0][i] = b0; acc[mt][1][i] = b1; }

    for (int kc = 0; kc < KK / CH; ++kc) {
        // ---- stage A chunk [128 x 128] into LDS as bf16 (coalesced 16B/8B) ----
        if (A_F32) {
            const float* A = (const float*)Aptr;
#pragma unroll
            for (int it = 0; it < (ROWS * CH) / (512 * 4); ++it) {  // 8 iters
                const int e = (it * 512 + tid) * 4;
                const int r = e / CH, c = e % CH;
                float4 v = *(const float4*)&A[(size_t)(row0 + r) * KK + kc * CH + c];
                US4 o; o.x = f2bf(v.x); o.y = f2bf(v.y); o.z = f2bf(v.z); o.w = f2bf(v.w);
                *(US4*)&As[r * LDA + c] = o;
                if (kc + 1 < KK / CH)   // warm L2/L1 for next K-chunk
                    __builtin_prefetch(&A[(size_t)(row0 + r) * KK + (kc + 1) * CH + c], 0, 1);
            }
        } else {
            const unsigned short* A = (const unsigned short*)Aptr;
#pragma unroll
            for (int it = 0; it < (ROWS * CH) / (512 * 4); ++it) {
                const int e = (it * 512 + tid) * 4;
                const int r = e / CH, c = e % CH;
                *(US4*)&As[r * LDA + c] =
                    *(const US4*)&A[(size_t)(row0 + r) * KK + kc * CH + c];
                if (kc + 1 < KK / CH)
                    __builtin_prefetch(&A[(size_t)(row0 + r) * KK + (kc + 1) * CH + c], 0, 1);
            }
        }
        __syncthreads();

#pragma unroll
        for (int kt = 0; kt < CH / 32; ++kt) {
            const int kb = kc * CH + kt * 32;
            // B fragments (32x16): per-lane N = l, contiguous K = hi*16 .. +15
            union BF { uint4 q[2]; v16bf v; } fb0, fb1;
            {
                const unsigned short* p0 = &W[(size_t)(colb + l) * KK + kb + hi * 16];
                fb0.q[0] = *(const uint4*)p0;       fb0.q[1] = *(const uint4*)(p0 + 8);
                const unsigned short* p1 = &W[(size_t)(colb + 16 + l) * KK + kb + hi * 16];
                fb1.q[0] = *(const uint4*)p1;       fb1.q[1] = *(const uint4*)(p1 + 8);
            }
            // A fragments (16x32): per-lane M = l, contiguous K = hi*16 .. +15.
            // Load ALL 8 up front: one ds_load clause, single wait, then an
            // uninterrupted 16-deep WMMA chain (hides LDS latency).
            union AF { uint4 q[2]; v16bf v; } fa[8];
#pragma unroll
            for (int mt = 0; mt < 8; ++mt) {
                const unsigned short* pa = &As[(mt * 16 + l) * LDA + kt * 32 + hi * 16];
                fa[mt].q[0] = *(const uint4*)pa;
                fa[mt].q[1] = *(const uint4*)(pa + 8);
            }
#pragma unroll
            for (int mt = 0; mt < 8; ++mt) {
                acc[mt][0] = __builtin_amdgcn_wmma_f32_16x16x32_bf16(
                    false, fa[mt].v, false, fb0.v, (short)0, acc[mt][0], false, false);
                acc[mt][1] = __builtin_amdgcn_wmma_f32_16x16x32_bf16(
                    false, fa[mt].v, false, fb1.v, (short)0, acc[mt][1], false, false);
            }
        }
        __syncthreads();
    }

    // ---- store: C layout -> lane holds N = colb+nt*16+l, vgpr i -> M = i+hi*8 ----
#pragma unroll
    for (int mt = 0; mt < 8; ++mt)
#pragma unroll
        for (int nt = 0; nt < 2; ++nt) {
            const int col = colb + nt * 16 + l;
#pragma unroll
            for (int i = 0; i < 8; ++i) {
                const size_t row = (size_t)(row0 + mt * 16 + hi * 8 + i);
                if (OUT_F32) ((float*)Out)[row * NN + col] = acc[mt][nt][i];
                else ((unsigned short*)Out)[row * NN + col] = f2bf(acc[mt][nt][i]);
            }
        }
}

// ------------------- LIF scan (in-place on bf16 h buffer) -------------------
// v += (h - (v - V_RESET))/TAU ; spike = v>=VTH ; hard reset. TAU=2,VTH=1,VR=0.
__global__ void lif_scan(unsigned short* __restrict__ hs) {
    const int idx = blockIdx.x * blockDim.x + threadIdx.x;   // covers B*N
    constexpr int BN = BB * NN;
    unsigned short* p = hs + idx;
    float v = 0.0f;
#pragma unroll 4
    for (int t = 0; t < TT; ++t) {
        const float h = bf2f(p[(size_t)t * BN]);
        v = v + (h - v) * 0.5f;                 // charge (tau = 2, v_reset = 0)
        const bool fire = (v >= 1.0f);
        p[(size_t)t * BN] = fire ? (unsigned short)0x3F80 : (unsigned short)0; // bf16 1/0
        v = fire ? 0.0f : v;                    // hard reset
    }
}

// ---------------------------------------------------------------------------
extern "C" void kernel_launch(void* const* d_in, const int* in_sizes, int n_in,
                              void* d_out, int out_size, void* d_ws, size_t ws_size,
                              hipStream_t stream) {
    const float* x     = (const float*)d_in[0];   // [64][1024][512]
    const float* fc1_w = (const float*)d_in[1];   // [512][512]
    const float* fc1_b = (const float*)d_in[2];   // [512]
    const float* fc2_w = (const float*)d_in[3];   // [512][512]
    const float* fc2_b = (const float*)d_in[4];   // [512]
    float* out = (float*)d_out;

    unsigned short* w1b  = (unsigned short*)d_ws;          // 512 KB
    unsigned short* w2b  = w1b + (size_t)NN * KK;          // 512 KB
    unsigned short* hbuf = w2b + (size_t)NN * KK;          // 64 MB bf16 h/spikes

    cvt_weights<<<(NN * KK) / 256, 256, 0, stream>>>(fc1_w, fc2_w, w1b, w2b);

    // Phase 1: h = x @ W1^T + b1   (fp32 A -> bf16 out, f32 WMMA accumulate)
    snn_gemm<true, false><<<MROW / ROWS, 512, 0, stream>>>(x, w1b, fc1_b, hbuf);

    // Phase 2: LIF membrane scan, spikes overwrite h in place (bf16 0/1)
    lif_scan<<<(BB * NN) / 256, 256, 0, stream>>>(hbuf);

    // Phase 3: out = s @ W2^T + b2  (bf16 A -> fp32 out)
    snn_gemm<false, true><<<MROW / ROWS, 512, 0, stream>>>(hbuf, w2b, fc2_b, out);
}
